// AttentionModelLargeMemory_37151467110735
// MI455X (gfx1250) — compile-verified, matrix operand order checked by reading, not run on hardware
//
#include <hip/hip_runtime.h>
#include <hip/hip_bf16.h>

// Problem constants
#define NB 32
#define NL 2048
#define NC 512
#define NQ 512
#define NH 1024

#define TILE_L 32               // L rows per block (2 M tiles)
#define HCOLS 128               // H columns per hIter per block (4 wave-pairs x 32)
#define KSLICE 64               // K window staged per LDS slice
#define COL_STRIDE 72           // ushorts per staged column (64 data + 8 pad = 144 B)
#define BUFE (HCOLS * COL_STRIDE) // 9216 ushorts = 18432 B per buffer

typedef __attribute__((ext_vector_type(16))) __bf16 v16bf;
typedef __attribute__((ext_vector_type(8)))  float  v8f;

union Frag {
    uint4  u[2];
    v16bf  v;
};

__device__ __forceinline__ unsigned int f2bf(float x) {
    unsigned int u = __float_as_uint(x);
    // round-to-nearest-even fp32 -> bf16
    return (u + 0x7FFFu + ((u >> 16) & 1u)) >> 16;
}

// pack two fp32 into packed bf16x2 (one v_cvt_pk_bf16_f32 when available)
#if __has_builtin(__builtin_amdgcn_cvt_pk_bf16_f32)
typedef __attribute__((ext_vector_type(2))) __bf16 bf16x2;
__device__ __forceinline__ unsigned int packpair(float x, float y) {
    bf16x2 p = __builtin_amdgcn_cvt_pk_bf16_f32(x, y);
    unsigned int r;
    __builtin_memcpy(&r, &p, 4);
    return r;
}
#else
__device__ __forceinline__ unsigned int packpair(float x, float y) {
    return f2bf(x) | (f2bf(y) << 16);
}
#endif

__device__ __forceinline__ uint4 pack8(float4 a, float4 b) {
    uint4 r;
    r.x = packpair(a.x, a.y);
    r.y = packpair(a.z, a.w);
    r.z = packpair(b.x, b.y);
    r.w = packpair(b.z, b.w);
    return r;
}

// branchless fast tanh: 1 - 2/(1+exp(2x)); exact limits at +/-inf
__device__ __forceinline__ float fast_tanh(float x) {
    return 1.0f - 2.0f * __builtin_amdgcn_rcpf(1.0f + __expf(2.0f * x));
}

// ---------------------------------------------------------------- prep: Wc->bf16, zero sums
__global__ __launch_bounds__(256) void k_prep(const float* __restrict__ Wc,
                                              unsigned int* __restrict__ WcBfPk,
                                              float* __restrict__ sums) {
    int idx = blockIdx.x * 256 + threadIdx.x;     // grid sized exactly NH*NC/2 = 262144
    if (idx < NB) sums[idx] = 0.0f;
    float2 v = ((const float2*)Wc)[idx];
    WcBfPk[idx] = packpair(v.x, v.y);
}

// ---------------------------------------------------------------- res_q = query @ Wq^T  [B,H]
__global__ __launch_bounds__(256) void k_resq(const float* __restrict__ query,
                                              const float* __restrict__ Wq,
                                              float* __restrict__ resq) {
    int idx = blockIdx.x * 256 + threadIdx.x;     // 0 .. NB*NH-1
    int b = idx >> 10;
    int h = idx & (NH - 1);
    const float4* q = (const float4*)(query + (size_t)b * NQ);
    const float4* w = (const float4*)(Wq + (size_t)h * NQ);
    float s = 0.0f;
    #pragma unroll 8
    for (int i = 0; i < NQ / 4; ++i) {
        float4 a = q[i], c = w[i];
        s += a.x * c.x + a.y * c.y + a.z * c.z + a.w * c.w;
    }
    resq[idx] = s;
}

// ---------------------------------------------------------------- fused GEMM+tanh+dot+exp
// grid = (NL/TILE_L, NB), block = 256 (8 waves).
// A (context rows, bf16) lives in registers: 16 fragments per wave, K=512.
// B (Wc) is double-buffered through LDS in 64-K slices shared by all waves.
__global__ __launch_bounds__(256) void k_main(const float* __restrict__ context,
                                              const float* __restrict__ mask,
                                              const unsigned short* __restrict__ WcBf,
                                              const float* __restrict__ resq,
                                              const float* __restrict__ bc,
                                              const float* __restrict__ Wo,
                                              const float* __restrict__ bo,
                                              float* __restrict__ weights_exp,
                                              float* __restrict__ sums) {
    __shared__ __align__(16) unsigned short ldsB[2 * BUFE];   // ~36.9 KB
    __shared__ float ldsLogit[TILE_L];
    __shared__ float ldsWe[TILE_L];

    const int b  = blockIdx.y;
    const int l0 = blockIdx.x * TILE_L;
    const int t  = threadIdx.x;
    const int lane  = t & 31;
    const int wave  = t >> 5;
    const int mtile = wave & 1;          // which 16-row M tile
    const int npair = wave >> 1;         // 0..3 -> 32-col H pair within the 128-col window
    const int lr    = lane & 15;
    const int hi    = lane >> 4;         // K half-group (ISA 16-bit A/B layout)

    if (t < TILE_L) ldsLogit[t] = 0.0f;

    // ---- A: load this wave's 16 context rows straight into registers (fp32->bf16) ----
    // lane (lr,hi) serves row l0+mtile*16+lr; holds K chunks {32f+hi*8..+7} & {32f+16+hi*8..+7}
    const float* arow = context + ((size_t)b * NL + l0 + mtile * 16 + lr) * NC;
    Frag aF[16];
    #pragma unroll
    for (int f = 0; f < 16; ++f) {
        const int kb = 32 * f + hi * 8;
        float4 p0 = *(const float4*)(arow + kb);
        float4 p1 = *(const float4*)(arow + kb + 4);
        float4 p2 = *(const float4*)(arow + kb + 16);
        float4 p3 = *(const float4*)(arow + kb + 20);
        aF[f].u[0] = pack8(p0, p1);
        aF[f].u[1] = pack8(p2, p3);
    }

    // ---- B staging: one 64-K slice of 128 columns -> 18 KB buffer ----
    // thread t stages 64 contiguous bytes: column t>>1, half t&1
    const int sCol  = t >> 1;
    const int sPart = t & 1;
    auto stageSlice = [&](int buf, int sl) {
        const int hI = sl >> 3, ss = sl & 7;
        const uint4* src = (const uint4*)(WcBf + (size_t)(hI * HCOLS + sCol) * NC
                                               + ss * KSLICE + sPart * 32);
        uint4* dst = (uint4*)(ldsB + (size_t)buf * BUFE + sCol * COL_STRIDE + sPart * 32);
        uint4 v0 = src[0], v1 = src[1], v2 = src[2], v3 = src[3];
        dst[0] = v0; dst[1] = v1; dst[2] = v2; dst[3] = v3;
    };

    float accA[8], accB[8];
    #pragma unroll
    for (int r = 0; r < 8; ++r) { accA[r] = 0.0f; accB[r] = 0.0f; }

    stageSlice(0, 0);
    __syncthreads();

    #pragma unroll 1
    for (int hIter = 0; hIter < NH / HCOLS; ++hIter) {      // 8 iterations
        v8f c0 = {}, c1 = {};
        #pragma unroll
        for (int s = 0; s < 8; ++s) {                        // 8 x 64-K slices
            const int sl = hIter * 8 + s;
            if (s < 7) {
                stageSlice((s + 1) & 1, sl + 1);             // prefetch next slice
            } else if (hIter < 7) {
                stageSlice((s + 1) & 1, sl + 1);             // first slice of next hIter
            }
            // consume slice s from buffer (s&1): load all 4 fragments first so the
            // in-order DS returns let WMMAs proceed on partial dscnt waits
            const unsigned short* colA = ldsB + (size_t)(s & 1) * BUFE
                                              + (npair * 32 + lr) * COL_STRIDE;
            const unsigned short* colB = colA + 16 * COL_STRIDE;
            const int kb0 = hi * 8;
            const int kb1 = 32 + hi * 8;
            Frag b00, b10, b01, b11;
            b00.u[0] = *(const uint4*)(colA + kb0);          // ds_load_b128 x8
            b00.u[1] = *(const uint4*)(colA + kb0 + 16);
            b10.u[0] = *(const uint4*)(colB + kb0);
            b10.u[1] = *(const uint4*)(colB + kb0 + 16);
            b01.u[0] = *(const uint4*)(colA + kb1);
            b01.u[1] = *(const uint4*)(colA + kb1 + 16);
            b11.u[0] = *(const uint4*)(colB + kb1);
            b11.u[1] = *(const uint4*)(colB + kb1 + 16);
            c0 = __builtin_amdgcn_wmma_f32_16x16x32_bf16(
                     false, aF[2 * s].v,     false, b00.v, (short)0, c0, false, false);
            c1 = __builtin_amdgcn_wmma_f32_16x16x32_bf16(
                     false, aF[2 * s].v,     false, b10.v, (short)0, c1, false, false);
            c0 = __builtin_amdgcn_wmma_f32_16x16x32_bf16(
                     false, aF[2 * s + 1].v, false, b01.v, (short)0, c0, false, false);
            c1 = __builtin_amdgcn_wmma_f32_16x16x32_bf16(
                     false, aF[2 * s + 1].v, false, b11.v, (short)0, c1, false, false);
            if (s == 7) {
                // epilogue for this 128-col H window: tanh(res_q+res_c+bc)*Wo per row
                const int hA = hIter * HCOLS + npair * 32 + lr;
                const int hB = hA + 16;
                const float rqA = resq[b * NH + hA] + bc[hA];
                const float rqB = resq[b * NH + hB] + bc[hB];
                const float woA = Wo[hA];
                const float woB = Wo[hB];
                #pragma unroll
                for (int r = 0; r < 8; ++r) {
                    accA[r] += woA * fast_tanh(c0[r] + rqA);
                    accB[r] += woB * fast_tanh(c1[r] + rqB);
                }
            }
            __syncthreads();   // staged slice visible; consumed buffer reusable
        }
    }

    // both column tiles map to the same rows: combine, then reduce across the
    // 16-lane half group (xor 1,2,4,8 stays inside the group)
    #pragma unroll
    for (int r = 0; r < 8; ++r) {
        float v = accA[r] + accB[r];
        v += __shfl_xor(v, 1, 32);
        v += __shfl_xor(v, 2, 32);
        v += __shfl_xor(v, 4, 32);
        v += __shfl_xor(v, 8, 32);
        accA[r] = v;
    }
    if (lr == 0) {                       // lanes 0 and 16 hold the two row-octet sums
        #pragma unroll
        for (int r = 0; r < 8; ++r) {
            int rr = mtile * 16 + hi * 8 + r;   // C/D layout: m = r + 8*(lane>=16)
            atomicAdd(&ldsLogit[rr], accA[r]);
        }
    }
    __syncthreads();

    // masked exp of logits, per-batch partial sum
    if (t < TILE_L) {
        int l = l0 + t;
        float logit = ldsLogit[t] + bo[0];
        float we = mask[(size_t)b * NL + l] * __expf(logit);
        weights_exp[(size_t)b * NL + l] = we;
        ldsWe[t] = we;
    }
    __syncthreads();
    if (t == 0) {
        float s = 0.0f;
        #pragma unroll
        for (int i = 0; i < TILE_L; ++i) s += ldsWe[i];
        atomicAdd(&sums[b], s);
    }
}

// ---------------------------------------------------------------- normalize weights, zero output
__global__ __launch_bounds__(256) void k_weights(const float* __restrict__ weights_exp,
                                                 const float* __restrict__ sums,
                                                 float* __restrict__ out) {
    int idx = blockIdx.x * 256 + threadIdx.x;   // 0 .. NB*NL-1
    int b = idx >> 11;
    float w = weights_exp[idx] / (sums[b] + 1e-5f);
    out[NB * NC + idx] = w;                     // weights output region
    if (idx < NB * NC) out[idx] = 0.0f;         // zero the 'output' region for atomics
}

// ---------------------------------------------------------------- output[b,c] = sum_l w[b,l]*ctx[b,l,c]
// grid = (16 L-slices, NB); each slice covers 128 rows; float4-coalesced over C.
__global__ __launch_bounds__(256) void k_out(const float* __restrict__ context,
                                             const float* wsrc,   // = d_out (weights region)
                                             float* out) {
    const int b  = blockIdx.y;
    const int l0 = blockIdx.x * 128;
    const int t  = threadIdx.x;
    const int c4 = t & 127;                     // float4 column
    const int rh = t >> 7;                      // 0/1 row parity
    const float* w = wsrc + NB * NC + (size_t)b * NL + l0;
    float4 acc = make_float4(0.f, 0.f, 0.f, 0.f);
    for (int r = rh; r < 128; r += 2) {
        float ww = w[r];
        float4 v = ((const float4*)(context + ((size_t)b * NL + l0 + r) * NC))[c4];
        acc.x += ww * v.x; acc.y += ww * v.y; acc.z += ww * v.z; acc.w += ww * v.w;
    }
    float* o = out + (size_t)b * NC + c4 * 4;
    atomicAdd(o + 0, acc.x);
    atomicAdd(o + 1, acc.y);
    atomicAdd(o + 2, acc.z);
    atomicAdd(o + 3, acc.w);
}

extern "C" void kernel_launch(void* const* d_in, const int* in_sizes, int n_in,
                              void* d_out, int out_size, void* d_ws, size_t ws_size,
                              hipStream_t stream) {
    (void)in_sizes; (void)n_in; (void)out_size; (void)ws_size;
    const float* query   = (const float*)d_in[0];
    const float* context = (const float*)d_in[1];
    const float* mask    = (const float*)d_in[2];
    const float* Wq      = (const float*)d_in[3];
    const float* Wc      = (const float*)d_in[4];
    const float* bc      = (const float*)d_in[5];
    const float* Wo      = (const float*)d_in[6];
    const float* bo      = (const float*)d_in[7];
    float* out = (float*)d_out;

    // workspace layout
    char* ws = (char*)d_ws;
    unsigned short* WcBf = (unsigned short*)ws;                          // 1 MB   (NH*NC bf16)
    float* resq  = (float*)(ws + (size_t)NH * NC * 2);                   // 128 KB (NB*NH f32)
    float* wexp  = (float*)(ws + (size_t)NH * NC * 2 + NB * NH * 4);     // 256 KB (NB*NL f32)
    float* sums  = (float*)(ws + (size_t)NH * NC * 2 + NB * NH * 4
                               + (size_t)NB * NL * 4);                   // 128 B

    k_prep   <<<(NH * NC) / 512, 256, 0, stream>>>(Wc, (unsigned int*)WcBf, sums);
    k_resq   <<<(NB * NH) / 256, 256, 0, stream>>>(query, Wq, resq);
    k_main   <<<dim3(NL / TILE_L, NB), 256, 0, stream>>>(context, mask, WcBf, resq,
                                                         bc, Wo, bo, wexp, sums);
    k_weights<<<(NB * NL) / 256, 256, 0, stream>>>(wexp, sums, out);
    k_out    <<<dim3(NL / 128, NB), 256, 0, stream>>>(context, out, out);
}
